// SelfAttention_80126909874397
// MI455X (gfx1250) — matrix operand order repaired and checked by reading, not compiled, not run on hardware
//
#include <hip/hip_runtime.h>

// Problem sizes (fixed by the reference)
#define B_   4
#define S_   2048
#define D_   1024
#define H_   16
#define DK_  64
#define EPS_ 1e-5f

typedef __attribute__((ext_vector_type(16))) _Float16 v16h;
typedef __attribute__((ext_vector_type(8)))  float    v8f;

union ABReg {
    v16h v;
    unsigned int u[8];
};

// ---------------------------------------------------------------------------
// Single-instruction DPP16 butterfly stages: VOP2 with DPP on src0 gives
//   dst = op(row_ror(x), x)
// in ONE VALU op -- no old-operand movs, no canonicalize, no bitcasts.
// A rotate butterfly over ror:8/4/2/1 leaves every lane of a 16-lane row
// holding the full row reduction; our score C-layout rows are exactly the
// 16-lane DPP rows, so this works per half-wave with no DS traffic.
// ---------------------------------------------------------------------------
#define DPP_STAGE(name, opstr, n)                                          \
__device__ __forceinline__ float name(float x) {                           \
    float d;                                                               \
    asm(opstr " %0, %1, %2 row_ror:" #n " row_mask:0xf bank_mask:0xf"      \
        : "=v"(d) : "v"(x), "v"(x));                                       \
    return d;                                                              \
}
DPP_STAGE(max_r8, "v_max_num_f32_dpp", 8)
DPP_STAGE(max_r4, "v_max_num_f32_dpp", 4)
DPP_STAGE(max_r2, "v_max_num_f32_dpp", 2)
DPP_STAGE(max_r1, "v_max_num_f32_dpp", 1)
DPP_STAGE(add_r8, "v_add_f32_dpp", 8)
DPP_STAGE(add_r4, "v_add_f32_dpp", 4)
DPP_STAGE(add_r2, "v_add_f32_dpp", 2)
DPP_STAGE(add_r1, "v_add_f32_dpp", 1)

__device__ __forceinline__ float row16_max(float x) {
    return max_r1(max_r2(max_r4(max_r8(x))));
}
__device__ __forceinline__ float row16_sum(float x) {
    return add_r1(add_r2(add_r4(add_r8(x))));
}

// ---------------------------------------------------------------------------
// Kernel 1: per-head QKV projection, f32 math, f16 outputs.
//   Q -> [bh][s][c], PRE-SCALED by 1/sqrt(DK) (folds softmax scale into GEMM)
//   K -> [bh][s][c]   (channel-contiguous: dword-packed B operand for Q.K^T)
//   V -> [bh][c][s]   (key-contiguous:     dword-packed B operand for P.V)
// grid = B*H * (S/16) blocks of 256 threads; each block does 16 rows of 1 head
// ---------------------------------------------------------------------------
__global__ __launch_bounds__(256) void qkv_proj_kernel(
    const float* __restrict__ x,
    const float* __restrict__ Wq, const float* __restrict__ bq,
    const float* __restrict__ Wk, const float* __restrict__ bk,
    const float* __restrict__ Wv, const float* __restrict__ bv,
    _Float16* __restrict__ Qf, _Float16* __restrict__ Kf, _Float16* __restrict__ Vt)
{
    __shared__ float Wl[3][DK_][DK_];   // 48 KB
    __shared__ float xl[16][DK_];       // 4 KB
    __shared__ float bl[3][DK_];

    const int t     = threadIdx.x;
    const int bh    = blockIdx.x >> 7;   // 0..63  (b*H + h)
    const int stile = blockIdx.x & 127;  // 0..127 (S/16)
    const int b     = bh >> 4;
    const int hh    = bh & 15;
    const int s0    = stile * 16;

    for (int i = t; i < DK_ * DK_; i += 256) {
        (&Wl[0][0][0])[i] = Wq[hh * DK_ * DK_ + i];
        (&Wl[1][0][0])[i] = Wk[hh * DK_ * DK_ + i];
        (&Wl[2][0][0])[i] = Wv[hh * DK_ * DK_ + i];
    }
    if (t < DK_) {
        bl[0][t] = bq[hh * DK_ + t];
        bl[1][t] = bk[hh * DK_ + t];
        bl[2][t] = bv[hh * DK_ + t];
    }
    for (int i = t; i < 16 * DK_; i += 256) {
        int r = i >> 6, k = i & 63;
        xl[r][k] = x[((size_t)(b * S_ + s0 + r)) * D_ + hh * DK_ + k];
    }
    __syncthreads();

    const int c  = t & 63;
    const int rg = t >> 6;
    for (int r = rg; r < 16; r += 4) {
        float qa = bl[0][c], ka = bl[1][c], va = bl[2][c];
        #pragma unroll 16
        for (int kk = 0; kk < DK_; ++kk) {
            float xv = xl[r][kk];
            qa = fmaf(xv, Wl[0][kk][c], qa);
            ka = fmaf(xv, Wl[1][kk][c], ka);
            va = fmaf(xv, Wl[2][kk][c], va);
        }
        const int s = s0 + r;
        const size_t qi = ((size_t)bh * S_ + s) * DK_ + c;
        Qf[qi] = (_Float16)(qa * 0.125f);       // fold 1/sqrt(64) into Q
        Kf[qi] = (_Float16)ka;
        Vt[((size_t)bh * DK_ + c) * S_ + s] = (_Float16)va;
    }
}

// ---------------------------------------------------------------------------
// Kernel 2: flash attention with v_wmma_f32_16x16x32_f16.
// grid = B*H * (S/64) blocks of 128 threads (4 waves); 1 wave = 16 query rows.
// ---------------------------------------------------------------------------
__global__ __launch_bounds__(128) void flash_attn_kernel(
    const _Float16* __restrict__ Qf, const _Float16* __restrict__ Kf,
    const _Float16* __restrict__ Vt, float* __restrict__ Out)
{
    __shared__ _Float16 Pl[4][16][32];  // per-wave P relayout buffer (C -> A)

    const int tid  = threadIdx.x;
    const int wave = tid >> 5;
    const int lane = tid & 31;
    const int hlf  = lane >> 4;   // half-wave: selects K range in A/B layouts
    const int l15  = lane & 15;

    const int bh    = blockIdx.x >> 5;  // 0..63
    const int qt    = blockIdx.x & 31;  // 0..31 (S/64)
    const int b     = bh >> 4;
    const int hh    = bh & 15;
    const int qbase = qt * 64 + wave * 16;

    const _Float16* Qp = Qf + ((size_t)bh * S_ + qbase) * DK_;
    const _Float16* Kp = Kf + (size_t)bh * S_ * DK_;
    const _Float16* Vp = Vt + (size_t)bh * DK_ * S_;

    // --- load Q tile as two A operands (16x32 f16 each), kept all kernel ---
    // A layout: lane -> row m = l15; VGPR v holds K pair at
    //           k0 = hlf*16 + v*2
    ABReg aq0, aq1;
    #pragma unroll
    for (int v = 0; v < 8; ++v) {
        const int c0 = (hlf << 4) + (v << 1);
        aq0.u[v] = *(const unsigned int*)(Qp + (size_t)l15 * DK_ + c0);
        aq1.u[v] = *(const unsigned int*)(Qp + (size_t)l15 * DK_ + c0 + 32);
    }

    v8f acc0 = {}, acc1 = {}, acc2 = {}, acc3 = {};
    float mrow[8], lrow[8];
    #pragma unroll
    for (int v = 0; v < 8; ++v) { mrow[v] = -3.0e38f; lrow[v] = 0.0f; }

    for (int kb = 0; kb < S_; kb += 32) {
        // prefetch next iteration's K rows into the cache hierarchy
        if (kb + 32 < S_) {
            __builtin_prefetch(Kp + (size_t)(kb + 32 + l15) * DK_, 0, 3);
            __builtin_prefetch(Kp + (size_t)(kb + 48 + l15) * DK_, 0, 3);
        }

        // ---- scores: two 16-key subtiles, each = 2 chained WMMAs over DK=64
        // B layout: lane -> col n = l15 (key); K = hlf*16 + 2v (+1) (channel)
        v8f s0 = {}, s1 = {};
        {
            ABReg bk0, bk1;
            const size_t krow = (size_t)(kb + l15) * DK_;
            #pragma unroll
            for (int v = 0; v < 8; ++v) {
                const int c0 = (hlf << 4) + (v << 1);
                bk0.u[v] = *(const unsigned int*)(Kp + krow + c0);
                bk1.u[v] = *(const unsigned int*)(Kp + krow + c0 + 32);
            }
            s0 = __builtin_amdgcn_wmma_f32_16x16x32_f16(false, aq0.v, false, bk0.v, (short)0, s0, false, false);
            s0 = __builtin_amdgcn_wmma_f32_16x16x32_f16(false, aq1.v, false, bk1.v, (short)0, s0, false, false);
        }
        {
            ABReg bk0, bk1;
            const size_t krow = (size_t)(kb + 16 + l15) * DK_;
            #pragma unroll
            for (int v = 0; v < 8; ++v) {
                const int c0 = (hlf << 4) + (v << 1);
                bk0.u[v] = *(const unsigned int*)(Kp + krow + c0);
                bk1.u[v] = *(const unsigned int*)(Kp + krow + c0 + 32);
            }
            s1 = __builtin_amdgcn_wmma_f32_16x16x32_f16(false, aq0.v, false, bk0.v, (short)0, s1, false, false);
            s1 = __builtin_amdgcn_wmma_f32_16x16x32_f16(false, aq1.v, false, bk1.v, (short)0, s1, false, false);
        }

        // ---- online softmax; C layout: row m = hlf*8 + v lives in 16 lanes.
        // Row reductions: single-instruction DPP row_ror butterflies.
        #pragma unroll
        for (int v = 0; v < 8; ++v) {
            const float a = s0[v];             // Q pre-scaled by 1/sqrt(DK)
            const float c = s1[v];
            const float mx   = row16_max(fmaxf(a, c));
            const float mnew = fmaxf(mrow[v], mx);
            const float corr = __expf(mrow[v] - mnew);
            const float p0 = __expf(a - mnew);
            const float p1 = __expf(c - mnew);
            const float rs = row16_sum(p0 + p1);
            lrow[v] = lrow[v] * corr + rs;
            mrow[v] = mnew;
            acc0[v] *= corr; acc1[v] *= corr; acc2[v] *= corr; acc3[v] *= corr;
            const int r = (hlf << 3) + v;
            Pl[wave][r][l15]      = (_Float16)p0;
            Pl[wave][r][16 + l15] = (_Float16)p1;
        }

        // per-wave LDS roundtrip: C layout -> A layout. Same-wave DS ops are
        // in-order; wait for stores + block compiler reordering explicitly.
        asm volatile("s_wait_dscnt 0" ::: "memory");

        ABReg ap;
        #pragma unroll
        for (int v = 0; v < 8; ++v) {
            const int k0 = (hlf << 4) + (v << 1);
            ap.u[v] = *(const unsigned int*)(&Pl[wave][l15][k0]);
        }

        // ---- P.V : B operand from Vt (keys contiguous); 4 channel tiles
        ABReg bv0, bv1, bv2, bv3;
        #pragma unroll
        for (int v = 0; v < 8; ++v) {
            const int key0 = kb + (hlf << 4) + (v << 1);
            bv0.u[v] = *(const unsigned int*)(Vp + (size_t)(l15)      * S_ + key0);
            bv1.u[v] = *(const unsigned int*)(Vp + (size_t)(16 + l15) * S_ + key0);
            bv2.u[v] = *(const unsigned int*)(Vp + (size_t)(32 + l15) * S_ + key0);
            bv3.u[v] = *(const unsigned int*)(Vp + (size_t)(48 + l15) * S_ + key0);
        }
        acc0 = __builtin_amdgcn_wmma_f32_16x16x32_f16(false, ap.v, false, bv0.v, (short)0, acc0, false, false);
        acc1 = __builtin_amdgcn_wmma_f32_16x16x32_f16(false, ap.v, false, bv1.v, (short)0, acc1, false, false);
        acc2 = __builtin_amdgcn_wmma_f32_16x16x32_f16(false, ap.v, false, bv2.v, (short)0, acc2, false, false);
        acc3 = __builtin_amdgcn_wmma_f32_16x16x32_f16(false, ap.v, false, bv3.v, (short)0, acc3, false, false);

        asm volatile("" ::: "memory"); // keep next-iter LDS stores behind reads
    }

    // ---- epilogue: normalize by l, scatter to [B,S,D] (pre-LayerNorm)
    #pragma unroll
    for (int v = 0; v < 8; ++v) {
        const int q = qbase + (hlf << 3) + v;
        const float inv = 1.0f / lrow[v];
        const size_t base = ((size_t)b * S_ + q) * D_ + hh * DK_ + l15;
        Out[base +  0] = acc0[v] * inv;
        Out[base + 16] = acc1[v] * inv;
        Out[base + 32] = acc2[v] * inv;
        Out[base + 48] = acc3[v] * inv;
    }
}

// ---------------------------------------------------------------------------
// Kernel 3: LayerNorm over D=1024, in place on d_out. grid = B*S rows.
// ---------------------------------------------------------------------------
__global__ __launch_bounds__(256) void layernorm_kernel(
    float* __restrict__ io, const float* __restrict__ gamma,
    const float* __restrict__ beta)
{
    __shared__ float red[256];
    const int t = threadIdx.x;
    float* p = io + (size_t)blockIdx.x * D_;

    float v[4];
    float s = 0.0f;
    #pragma unroll
    for (int i = 0; i < 4; ++i) { v[i] = p[t + i * 256]; s += v[i]; }
    red[t] = s;
    __syncthreads();
    for (int off = 128; off > 0; off >>= 1) {
        if (t < off) red[t] += red[t + off];
        __syncthreads();
    }
    const float mu = red[0] * (1.0f / D_);
    __syncthreads();

    s = 0.0f;
    #pragma unroll
    for (int i = 0; i < 4; ++i) { const float d = v[i] - mu; s += d * d; }
    red[t] = s;
    __syncthreads();
    for (int off = 128; off > 0; off >>= 1) {
        if (t < off) red[t] += red[t + off];
        __syncthreads();
    }
    const float rstd = rsqrtf(red[0] * (1.0f / D_) + EPS_);

    #pragma unroll
    for (int i = 0; i < 4; ++i) {
        const int c = t + i * 256;
        p[c] = (v[i] - mu) * rstd * gamma[c] + beta[c];
    }
}

// ---------------------------------------------------------------------------
extern "C" void kernel_launch(void* const* d_in, const int* in_sizes, int n_in,
                              void* d_out, int out_size, void* d_ws, size_t ws_size,
                              hipStream_t stream)
{
    const float* x     = (const float*)d_in[0];
    const float* Wq    = (const float*)d_in[1];
    const float* bq    = (const float*)d_in[2];
    const float* Wk    = (const float*)d_in[3];
    const float* bk    = (const float*)d_in[4];
    const float* Wv    = (const float*)d_in[5];
    const float* bv    = (const float*)d_in[6];
    const float* gamma = (const float*)d_in[7];
    const float* beta  = (const float*)d_in[8];
    float* out = (float*)d_out;

    const size_t nQKV = (size_t)B_ * H_ * S_ * DK_;   // 8.4M elems, f16
    _Float16* Qf = (_Float16*)d_ws;                   // 16 MB
    _Float16* Kf = Qf + nQKV;                         // 16 MB
    _Float16* Vt = Kf + nQKV;                         // 16 MB (transposed)

    qkv_proj_kernel<<<B_ * H_ * (S_ / 16), 256, 0, stream>>>(
        x, Wq, bq, Wk, bk, Wv, bv, Qf, Kf, Vt);
    flash_attn_kernel<<<B_ * H_ * (S_ / 64), 128, 0, stream>>>(Qf, Kf, Vt, out);
    layernorm_kernel<<<B_ * S_, 256, 0, stream>>>(out, gamma, beta);
}
